// ProjectedAdaptiveLogSoftmax_89266600280362
// MI455X (gfx1250) — matrix operand order repaired and checked by reading, not compiled, not run on hardware
//
#include <hip/hip_runtime.h>
#include <hip/hip_bf16.h>

// ---------------------------------------------------------------------------
// Projected adaptive log-softmax for MI455X (gfx1250), bf16 WMMA path.
// Round 3: pre-converted bf16 weights (pure load->wmma hot loop, gated on
// ws_size), 64-row LDS A tile (4x less L2 weight traffic), dynamic LDS,
// software-pipelined double-buffered B prefetch, dual column subtiles/wave.
// ---------------------------------------------------------------------------

typedef __attribute__((ext_vector_type(16))) __bf16          v16bf;
typedef __attribute__((ext_vector_type(8)))  float           v8f;
typedef __attribute__((ext_vector_type(16))) unsigned short  v16us;

struct alignas(16) U128 { unsigned int x, y, z, w; };

union Frag16 {
  v16us u;
  v16bf b;
  U128  q[2];
};

__device__ __forceinline__ unsigned short f2bf(float f) {
  union { float f; unsigned int u; } cv; cv.f = f;
  unsigned int u = cv.u;
  return (unsigned short)((u + 0x7fffu + ((u >> 16) & 1u)) >> 16);  // RNE
}

// --------------------------- elementwise converts ---------------------------

__global__ void cvt_bf16_kernel(const float* __restrict__ in,
                                unsigned short* __restrict__ out, int n) {
  int i = blockIdx.x * blockDim.x + threadIdx.x;
  if (i < n) out[i] = f2bf(in[i]);
}

// Row-major [rows, Kin] f32 -> [rows, Kout] bf16 with zero padding Kin..Kout.
__global__ void cvt_pad_bf16_kernel(const float* __restrict__ in,
                                    unsigned short* __restrict__ out,
                                    int Kin, int Kout, long total) {
  long i = (long)blockIdx.x * blockDim.x + threadIdx.x;
  if (i < total) {
    long row = i / Kout;
    int  k   = (int)(i - row * Kout);
    out[i] = (k < Kin) ? f2bf(in[row * Kin + k]) : (unsigned short)0;
  }
}

// p is [K, E] row-major f32; pT is [E, K] bf16.
__global__ void tcvt_kernel(const float* __restrict__ p,
                            unsigned short* __restrict__ pT, int K, int E) {
  int i = blockIdx.x * blockDim.x + threadIdx.x;
  if (i < K * E) {
    int e = i / K;
    int k = i - e * K;
    pT[i] = f2bf(p[(size_t)k * E + e]);
  }
}

// ---------------------------------------------------------------------------
// Fused tiled GEMM (+ optional online softmax with target gather).
//   A    : [Nrows, KDIM] bf16 (row-major, KDIM multiple of 32)
//   B0/B1: vocab weights, row r = column r of logits; bf16 stride KDIM or
//          f32 stride Kvalid.  Rows >= split come from B1.
//   SOFTMAX=false: store C (bf16) to Cout with row stride outStride.
//   SOFTMAX=true : per-chunk online logsumexp partials + gathered logit.
// Block: 256 threads = 8 waves; block owns 64 rows (A in dynamic LDS,
// 64*KDIM*2 bytes).  Wave = (row tile 0..3) x (32-col strip 0..1) of a
// 64-wide column group; gridDim.y vocab chunks (strided).  B loads are
// double-buffered one K-step ahead to overlap with WMMA.
// ---------------------------------------------------------------------------
template <int KDIM, bool BF16B, bool SOFTMAX>
__launch_bounds__(256)
__global__ void adasoft_gemm(const unsigned short* __restrict__ A,
                             const void* __restrict__ B0,
                             const void* __restrict__ B1,
                             const float* __restrict__ bias0,
                             const float* __restrict__ bias1,
                             int split, int V, int Kvalid,
                             const int* __restrict__ target, int mode,
                             float* __restrict__ pm, float* __restrict__ ps,
                             float* __restrict__ pg,
                             unsigned short* __restrict__ Cout, int outStride,
                             int Nrows) {
  extern __shared__ __align__(16) unsigned short smA[];   // 64 x KDIM bf16
  __shared__ float smM[2][64];
  __shared__ float smS[2][64];
  __shared__ float smG[64];

  const int tid     = threadIdx.x;
  const int rowBase = blockIdx.x * 64;
  const int cz      = blockIdx.y;
  const int Cc      = gridDim.y;

  if (SOFTMAX && tid < 64) smG[tid] = -1e30f;

  // Stage A tile: 64 rows x KDIM bf16, contiguous in global.
  {
    const U128* src = (const U128*)(A + (size_t)rowBase * KDIM);
    U128* dst = (U128*)smA;
    const int n16 = 64 * KDIM / 8;
    for (int i = tid; i < n16; i += 256) dst[i] = src[i];
  }
  __syncthreads();

  const int lane = tid & 31;
  const int half = lane >> 4;        // wave32: half 0 = lanes 0-15
  const int l16  = lane & 15;
  const int wave = tid >> 5;
  const int rt   = wave >> 1;        // 0..3: 16-row tile
  const int cs   = wave & 1;         // 0..1: 32-col strip within 64-col group
  const int rowTile = rowBase + rt * 16;

  // Per-row gather column (target logit / head cluster column).
  int gcol[8];
  if (SOFTMAX) {
#pragma unroll
    for (int r = 0; r < 8; ++r) {
      int row = rowTile + r + 8 * half;
      int t = target[row];
      int gc;
      if (mode == 0) {
        // head: target if shortlist, else cluster column (quirk: -1,-2,-3)
        gc = (t < 20000) ? t
                         : ((t < 40000) ? 20002 : ((t < 200000) ? 20001 : 20000));
      } else {
        int lo = (mode == 1) ? 20000 : ((mode == 2) ? 40000 : 200000);
        int sz = (mode == 1) ? 20000 : ((mode == 2) ? 160000 : 67735);
        int gi = t - lo;
        gc = gi < 0 ? 0 : (gi > sz - 1 ? sz - 1 : gi);
      }
      gcol[r] = gc;
    }
  }

  float m[8], s[8], g[8];
#pragma unroll
  for (int r = 0; r < 8; ++r) { m[r] = -1e30f; s[r] = 0.0f; g[r] = -1e30f; }

  const unsigned short* smRow = smA + (size_t)(rt * 16 + l16) * KDIM;
  const int nGroups = (V + 63) / 64;

  for (int cg = cz; cg < nGroups; cg += Cc) {
    const int colBase0 = cg * 64 + cs * 32;   // subtile 0
    const int colBase1 = colBase0 + 16;       // subtile 1
    const int col0 = colBase0 + l16;
    const int col1 = colBase1 + l16;
    const bool cv0 = col0 < V;
    const bool cv1 = col1 < V;
    const int bcol0 = cv0 ? col0 : 0;
    const int bcol1 = cv1 ? col1 : 0;

    // B row pointers (one per subtile).
    const unsigned short* brow0h = nullptr; const unsigned short* brow1h = nullptr;
    const float* brow0f = nullptr; const float* brow1f = nullptr;
    if (BF16B) {
      brow0h = (bcol0 < split) ? ((const unsigned short*)B0 + (size_t)bcol0 * KDIM)
                               : ((const unsigned short*)B1 + (size_t)(bcol0 - split) * KDIM);
      brow1h = (bcol1 < split) ? ((const unsigned short*)B0 + (size_t)bcol1 * KDIM)
                               : ((const unsigned short*)B1 + (size_t)(bcol1 - split) * KDIM);
    } else {
      brow0f = (bcol0 < split) ? ((const float*)B0 + (size_t)bcol0 * Kvalid)
                               : ((const float*)B1 + (size_t)(bcol0 - split) * Kvalid);
      brow1f = (bcol1 < split) ? ((const float*)B0 + (size_t)bcol1 * Kvalid)
                               : ((const float*)B1 + (size_t)(bcol1 - split) * Kvalid);
    }

    // Double-buffered raw B data (prefetched one K-step ahead).
    U128   braw[2][2][2];   // [parity][subtile][16B chunk]     (bf16 path)
    float4 fraw[2][2][4];   // [parity][subtile][4-float chunk] (f32 path)

    auto fetchB = [&](int kb, int par) {
      if (BF16B) {
        const unsigned short* r0 = brow0h + kb + 16 * half;
        const unsigned short* r1 = brow1h + kb + 16 * half;
        braw[par][0][0] = *(const U128*)(r0);
        braw[par][0][1] = *(const U128*)(r0 + 8);
        braw[par][1][0] = *(const U128*)(r1);
        braw[par][1][1] = *(const U128*)(r1 + 8);
      } else {
        const int k0 = kb + 16 * half;
#pragma unroll
        for (int jj = 0; jj < 4; ++jj) {
          const int kk = k0 + jj * 4;
          if (kk < Kvalid) {
            fraw[par][0][jj] = *(const float4*)(brow0f + kk);
            fraw[par][1][jj] = *(const float4*)(brow1f + kk);
          } else {
            fraw[par][0][jj] = make_float4(0.f, 0.f, 0.f, 0.f);
            fraw[par][1][jj] = make_float4(0.f, 0.f, 0.f, 0.f);
          }
        }
      }
    };

    auto buildB = [&](int par, int sub) {
      Frag16 b;
      if (BF16B) {
        b.q[0] = braw[par][sub][0];
        b.q[1] = braw[par][sub][1];
      } else {
#pragma unroll
        for (int jj = 0; jj < 4; ++jj) {
          float4 f = fraw[par][sub][jj];
          b.u[jj * 4 + 0] = f2bf(f.x);
          b.u[jj * 4 + 1] = f2bf(f.y);
          b.u[jj * 4 + 2] = f2bf(f.z);
          b.u[jj * 4 + 3] = f2bf(f.w);
        }
      }
      return b;
    };

    v8f c0 = {};
    v8f c1 = {};

    fetchB(0, 0);
#pragma unroll 2
    for (int kb = 0; kb < KDIM; kb += 32) {
      const int par = (kb >> 5) & 1;
      if (kb + 32 < KDIM) fetchB(kb + 32, par ^ 1);   // prefetch next step
      Frag16 a;
      a.q[0] = *(const U128*)(smRow + kb + 16 * half);      // K kb..kb+7 (+16 half1)
      a.q[1] = *(const U128*)(smRow + kb + 16 * half + 8);  // K kb+8..   (+16 half1)
      Frag16 b0 = buildB(par, 0);
      Frag16 b1 = buildB(par, 1);
      c0 = __builtin_amdgcn_wmma_f32_16x16x32_bf16(false, a.b, false, b0.b,
                                                   (short)0, c0, false, false);
      c1 = __builtin_amdgcn_wmma_f32_16x16x32_bf16(false, a.b, false, b1.b,
                                                   (short)0, c1, false, false);
    }

    if (SOFTMAX) {
      auto update = [&](const v8f& c, int colBase, bool cv, int col) {
        const float bias =
            cv ? ((col < split) ? bias0[col] : bias1[col - split]) : 0.0f;
#pragma unroll
        for (int r = 0; r < 8; ++r) {
          float v = cv ? (c[r] + bias) : -1e30f;
          float tm = v;   // row max across the 16 lanes of this half
#pragma unroll
          for (int off = 1; off < 16; off <<= 1)
            tm = fmaxf(tm, __shfl_xor(tm, off, 32));
          const float nm = fmaxf(m[r], tm);
          float p = __expf(v - nm);
#pragma unroll
          for (int off = 1; off < 16; off <<= 1) p += __shfl_xor(p, off, 32);
          s[r] = s[r] * __expf(m[r] - nm) + p;
          m[r] = nm;
          const int gc = gcol[r];
          if (gc >= colBase && gc < colBase + 16) {
            g[r] = __shfl(v, half * 16 + (gc - colBase), 32);
          }
        }
      };
      update(c0, colBase0, cv0, col0);
      update(c1, colBase1, cv1, col1);
    } else {
#pragma unroll
      for (int r = 0; r < 8; ++r) {
        const int row = rowTile + r + 8 * half;
        if (col0 < outStride)
          Cout[(size_t)row * outStride + col0] =
              cv0 ? f2bf(c0[r]) : (unsigned short)0;
        if (col1 < outStride)
          Cout[(size_t)row * outStride + col1] =
              cv1 ? f2bf(c1[r]) : (unsigned short)0;
      }
    }
  }

  if (SOFTMAX) {
    if (l16 == 0) {
#pragma unroll
      for (int r = 0; r < 8; ++r) {
        const int rl = rt * 16 + r + 8 * half;
        smM[cs][rl] = m[r];
        smS[cs][rl] = s[r];
      }
#pragma unroll
      for (int r = 0; r < 8; ++r) {
        const int gc = gcol[r];
        // owner: 32-col strip within 64-group, and vocab chunk
        if (((gc >> 5) & 1) == cs && ((gc >> 6) % Cc) == cz)
          smG[rt * 16 + r + 8 * half] = g[r];
      }
    }
    __syncthreads();
    if (tid < 64) {
      const float m0 = smM[0][tid], m1 = smM[1][tid];
      const float M = fmaxf(m0, m1);
      const float S = smS[0][tid] * __expf(m0 - M) + smS[1][tid] * __expf(m1 - M);
      const size_t idx = (size_t)cz * Nrows + rowBase + tid;
      pm[idx] = M;
      ps[idx] = S;
      pg[idx] = smG[tid];
    }
  }
}

// Merge vocab-chunk partials: contrib[row] = gathered_logit - logsumexp(row)
__global__ void reduce_kernel(const float* __restrict__ pm,
                              const float* __restrict__ ps,
                              const float* __restrict__ pg,
                              int Cc, int Nrows, float* __restrict__ contrib) {
  int row = blockIdx.x * blockDim.x + threadIdx.x;
  if (row >= Nrows) return;
  float M = -1e30f;
  for (int c = 0; c < Cc; ++c) M = fmaxf(M, pm[(size_t)c * Nrows + row]);
  float S = 0.0f, G = -1e30f;
  for (int c = 0; c < Cc; ++c) {
    S += ps[(size_t)c * Nrows + row] * __expf(pm[(size_t)c * Nrows + row] - M);
    G = fmaxf(G, pg[(size_t)c * Nrows + row]);
  }
  contrib[row] = G - (M + __logf(S));
}

__global__ void combine_kernel(const int* __restrict__ target,
                               const float* __restrict__ hc,
                               const float* __restrict__ c1,
                               const float* __restrict__ c2,
                               const float* __restrict__ c3,
                               float* __restrict__ out, int Nrows) {
  int row = blockIdx.x * blockDim.x + threadIdx.x;
  if (row >= Nrows) return;
  int t = target[row];
  float lp = hc[row];
  if (t >= 20000)
    lp += (t < 40000) ? c1[row] : ((t < 200000) ? c2[row] : c3[row]);
  out[row] = -lp;
}

// ---------------------------------------------------------------------------

extern "C" void kernel_launch(void* const* d_in, const int* in_sizes, int n_in,
                              void* d_out, int out_size, void* d_ws, size_t ws_size,
                              hipStream_t stream) {
  (void)in_sizes; (void)n_in; (void)out_size;

  const float* hidden    = (const float*)d_in[0];
  const int*   target    = (const int*)d_in[1];
  const float* w0        = (const float*)d_in[2];
  const float* b0        = (const float*)d_in[3];
  const float* cluster_w = (const float*)d_in[4];
  const float* cluster_b = (const float*)d_in[5];
  const float* w1        = (const float*)d_in[6];
  const float* b1        = (const float*)d_in[7];
  const float* w2        = (const float*)d_in[8];
  const float* b2        = (const float*)d_in[9];
  const float* w3        = (const float*)d_in[10];
  const float* b3        = (const float*)d_in[11];
  const float* p0        = (const float*)d_in[12];
  const float* p1        = (const float*)d_in[13];
  const float* p2        = (const float*)d_in[14];
  const float* p3        = (const float*)d_in[15];
  float* out = (float*)d_out;

  const int N = 1024, D = 1024;

  // Workspace carve-up (256B aligned). Base ~7.8 MB; bf16 weights ~76 MB
  // (optional, gated on ws_size).
  char* ws = (char*)d_ws;
  size_t off = 0;
  auto take = [&](size_t bytes) -> char* {
    char* r = ws + off;
    off += (bytes + 255) & ~(size_t)255;
    return r;
  };
  unsigned short* hB  = (unsigned short*)take((size_t)N * D * 2);
  unsigned short* pT0 = (unsigned short*)take((size_t)1024 * 1024 * 2);
  unsigned short* pT1 = (unsigned short*)take((size_t)256 * 1024 * 2);
  unsigned short* pT2 = (unsigned short*)take((size_t)64 * 1024 * 2);
  unsigned short* pT3 = (unsigned short*)take((size_t)16 * 1024 * 2);
  unsigned short* hp0 = (unsigned short*)take((size_t)N * 1024 * 2);
  unsigned short* hp1 = (unsigned short*)take((size_t)N * 256 * 2);
  unsigned short* hp2 = (unsigned short*)take((size_t)N * 64 * 2);
  unsigned short* hp3 = (unsigned short*)take((size_t)N * 32 * 2);   // K=16 padded to 32
  float* pm = (float*)take((size_t)16 * N * 4);
  float* ps = (float*)take((size_t)16 * N * 4);
  float* pg = (float*)take((size_t)16 * N * 4);
  float* hc  = (float*)take((size_t)N * 4);
  float* c1c = (float*)take((size_t)N * 4);
  float* c2c = (float*)take((size_t)N * 4);
  float* c3c = (float*)take((size_t)N * 4);
  // Optional pre-converted bf16 weights (w3 padded K=16 -> 32).
  unsigned short* w0b = (unsigned short*)take((size_t)20000 * 1024 * 2);
  unsigned short* cwb = (unsigned short*)take((size_t)3 * 1024 * 2);
  unsigned short* w1b = (unsigned short*)take((size_t)20000 * 256 * 2);
  unsigned short* w2b = (unsigned short*)take((size_t)160000 * 64 * 2);
  unsigned short* w3b = (unsigned short*)take((size_t)67735 * 32 * 2);
  const bool useWB = off <= ws_size;   // bf16-weight fast path fits?

  // 1) hidden -> bf16; projections -> transposed bf16 [E,K]
  cvt_bf16_kernel<<<dim3((N * D + 255) / 256), 256, 0, stream>>>(hidden, hB, N * D);
  tcvt_kernel<<<dim3((1024 * 1024 + 255) / 256), 256, 0, stream>>>(p0, pT0, 1024, 1024);
  tcvt_kernel<<<dim3((1024 * 256 + 255) / 256), 256, 0, stream>>>(p1, pT1, 1024, 256);
  tcvt_kernel<<<dim3((1024 * 64 + 255) / 256), 256, 0, stream>>>(p2, pT2, 1024, 64);
  tcvt_kernel<<<dim3((1024 * 16 + 255) / 256), 256, 0, stream>>>(p3, pT3, 1024, 16);

  if (useWB) {   // one-time weight conversion: O(V*K) instead of O(N*V*K)
    long t0 = (long)20000 * 1024, tc = (long)3 * 1024, t1 = (long)20000 * 256;
    long t2 = (long)160000 * 64, t3 = (long)67735 * 32;
    cvt_pad_bf16_kernel<<<dim3((unsigned)((t0 + 255) / 256)), 256, 0, stream>>>(w0, w0b, 1024, 1024, t0);
    cvt_pad_bf16_kernel<<<dim3((unsigned)((tc + 255) / 256)), 256, 0, stream>>>(cluster_w, cwb, 1024, 1024, tc);
    cvt_pad_bf16_kernel<<<dim3((unsigned)((t1 + 255) / 256)), 256, 0, stream>>>(w1, w1b, 256, 256, t1);
    cvt_pad_bf16_kernel<<<dim3((unsigned)((t2 + 255) / 256)), 256, 0, stream>>>(w2, w2b, 64, 64, t2);
    cvt_pad_bf16_kernel<<<dim3((unsigned)((t3 + 255) / 256)), 256, 0, stream>>>(w3, w3b, 16, 32, t3);
  }

  // 2) projection GEMMs: hp_i = hidden @ p_i  (bf16 WMMA, f32 accumulate)
  adasoft_gemm<1024, true, false><<<dim3(N / 64, 16), 256, 64 * 1024 * 2, stream>>>(
      hB, pT0, pT0, nullptr, nullptr, 1024, 1024, 1024, nullptr, 0,
      nullptr, nullptr, nullptr, hp0, 1024, N);
  adasoft_gemm<1024, true, false><<<dim3(N / 64, 4), 256, 64 * 1024 * 2, stream>>>(
      hB, pT1, pT1, nullptr, nullptr, 256, 256, 1024, nullptr, 0,
      nullptr, nullptr, nullptr, hp1, 256, N);
  adasoft_gemm<1024, true, false><<<dim3(N / 64, 1), 256, 64 * 1024 * 2, stream>>>(
      hB, pT2, pT2, nullptr, nullptr, 64, 64, 1024, nullptr, 0,
      nullptr, nullptr, nullptr, hp2, 64, N);
  adasoft_gemm<1024, true, false><<<dim3(N / 64, 1), 256, 64 * 1024 * 2, stream>>>(
      hB, pT3, pT3, nullptr, nullptr, 16, 16, 1024, nullptr, 0,
      nullptr, nullptr, nullptr, hp3, 32, N);

  // 3) fused logit GEMM + online softmax per cluster (partials), then reduce.
  if (useWB) {
    adasoft_gemm<1024, true, true><<<dim3(N / 64, 16), 256, 64 * 1024 * 2, stream>>>(
        hp0, w0b, cwb, b0, cluster_b, 20000, 20003, 1024, target, 0,
        pm, ps, pg, nullptr, 0, N);
    reduce_kernel<<<dim3(N / 256), 256, 0, stream>>>(pm, ps, pg, 16, N, hc);

    adasoft_gemm<256, true, true><<<dim3(N / 64, 16), 256, 64 * 256 * 2, stream>>>(
        hp1, w1b, w1b, b1, b1, 20000, 20000, 256, target, 1,
        pm, ps, pg, nullptr, 0, N);
    reduce_kernel<<<dim3(N / 256), 256, 0, stream>>>(pm, ps, pg, 16, N, c1c);

    adasoft_gemm<64, true, true><<<dim3(N / 64, 16), 256, 64 * 64 * 2, stream>>>(
        hp2, w2b, w2b, b2, b2, 160000, 160000, 64, target, 2,
        pm, ps, pg, nullptr, 0, N);
    reduce_kernel<<<dim3(N / 256), 256, 0, stream>>>(pm, ps, pg, 16, N, c2c);

    adasoft_gemm<32, true, true><<<dim3(N / 64, 16), 256, 64 * 32 * 2, stream>>>(
        hp3, w3b, w3b, b3, b3, 67735, 67735, 16, target, 3,
        pm, ps, pg, nullptr, 0, N);
    reduce_kernel<<<dim3(N / 256), 256, 0, stream>>>(pm, ps, pg, 16, N, c3c);
  } else {
    adasoft_gemm<1024, false, true><<<dim3(N / 64, 16), 256, 64 * 1024 * 2, stream>>>(
        hp0, w0, cluster_w, b0, cluster_b, 20000, 20003, 1024, target, 0,
        pm, ps, pg, nullptr, 0, N);
    reduce_kernel<<<dim3(N / 256), 256, 0, stream>>>(pm, ps, pg, 16, N, hc);

    adasoft_gemm<256, false, true><<<dim3(N / 64, 16), 256, 64 * 256 * 2, stream>>>(
        hp1, w1, w1, b1, b1, 20000, 20000, 256, target, 1,
        pm, ps, pg, nullptr, 0, N);
    reduce_kernel<<<dim3(N / 256), 256, 0, stream>>>(pm, ps, pg, 16, N, c1c);

    adasoft_gemm<64, false, true><<<dim3(N / 64, 16), 256, 64 * 64 * 2, stream>>>(
        hp2, w2, w2, b2, b2, 160000, 160000, 64, target, 2,
        pm, ps, pg, nullptr, 0, N);
    reduce_kernel<<<dim3(N / 256), 256, 0, stream>>>(pm, ps, pg, 16, N, c2c);

    adasoft_gemm<32, false, true><<<dim3(N / 64, 16), 256, 64 * 32 * 2, stream>>>(
        hp3, w3, w3, b3, b3, 67735, 67735, 16, target, 3,
        pm, ps, pg, nullptr, 0, N);
    reduce_kernel<<<dim3(N / 256), 256, 0, stream>>>(pm, ps, pg, 16, N, c3c);
  }

  // 4) cluster select + negate
  combine_kernel<<<dim3(N / 256), 256, 0, stream>>>(target, hc, c1c, c2c, c3c, out, N);
}
